// OuterLSTM_3994319585339
// MI455X (gfx1250) — matrix operand order repaired and checked by reading, not compiled, not run on hardware
//
#include <hip/hip_runtime.h>
#include <hip/hip_bf16.h>

#define SEQ 2048
#define EMB 1024
#define HID 1024
#define G4  4096   // 4*HID

typedef __attribute__((ext_vector_type(16))) __bf16 v16bf;
typedef __attribute__((ext_vector_type(8)))  float  v8f;
typedef __attribute__((ext_vector_type(8)))  __bf16 bf16x8;

// ---------------------------------------------------------------------------
// Phase 0: deterministic init of persistent state (h0 = 0, barrier = 0)
// ---------------------------------------------------------------------------
__global__ void lstm_init_kernel(float* __restrict__ h_glob,
                                 unsigned* __restrict__ bar) {
  int t = threadIdx.x;
  for (int i = t; i < HID; i += blockDim.x) h_glob[i] = 0.f;
  if (t == 0) { bar[0] = 0u; bar[1] = 0u; }
}

// ---------------------------------------------------------------------------
// Phase 1: gx_all[t] = emb_table[sentence[t]] @ Wx + bx   (M=2048,N=4096,K=1024)
// bf16 WMMA, f32 accumulate. 128x128 tile / WG, 8 waves, each wave 16 rows x
// 8 accumulators. Register double-buffered global->LDS staging, and a 2-deep
// software pipeline on B fragments so each WMMA overlaps the next ds_loads.
// ---------------------------------------------------------------------------
#define TM 128
#define TN 128
#define TK 32
#define LDA 40   // padded bf16 row length for A tile (32 K + pad)
#define LDB 40   // padded bf16 row length for B^T tile

__device__ __forceinline__ v16bf load_bfrag(const __bf16* __restrict__ Bls,
                                            int j, int l16, int half) {
  // B fragment: N = lane%16, K = e%8 + (e/8)*16 + half*8
  const __bf16* bp = &Bls[(j * 16 + l16) * LDB + half * 8];
  const bf16x8 b0 = *(const bf16x8*)(bp);
  const bf16x8 b1 = *(const bf16x8*)(bp + 16);
  v16bf b;
  #pragma unroll
  for (int e = 0; e < 8; ++e) { b[e] = b0[e]; b[8 + e] = b1[e]; }
  return b;
}

__global__ __launch_bounds__(256)
void gx_gemm_kernel(const int*   __restrict__ sent,
                    const float* __restrict__ emb,
                    const float* __restrict__ Wx,
                    const float* __restrict__ bx,
                    float*       __restrict__ gx) {
  __shared__ __bf16 Als[TM * LDA];   // [m][k]
  __shared__ __bf16 Bls[TN * LDB];   // transposed: [n][k]
  __shared__ int    tok[TM];

  const int t    = threadIdx.x;
  const int lane = t & 31;
  const int w    = t >> 5;        // wave 0..7
  const int half = lane >> 4;     // 0/1
  const int l16  = lane & 15;
  const int N0   = blockIdx.x * TN;
  const int M0   = blockIdx.y * TM;

  if (t < TM) tok[t] = sent[M0 + t];
  __syncthreads();

  // staging geometry (constant per thread)
  const int a_kg    = (t & 7) * 4;   // A: K group of 4
  const int a_rbase = t >> 3;        // A: row base 0..31
  const int b_cg    = (t & 31) * 4;  // B: col group of 4
  const int b_kk0   = t >> 5;        // B: k base 0..7

  v8f acc[8] = {};
  float4 aprf[4], bprf[4];

  // prologue: prefetch K-tile 0 into registers
  #pragma unroll
  for (int pass = 0; pass < 4; ++pass) {
    aprf[pass] = *(const float4*)(emb + (size_t)tok[a_rbase + pass * 32] * EMB + a_kg);
    bprf[pass] = *(const float4*)(Wx + (size_t)(b_kk0 + pass * 8) * G4 + N0 + b_cg);
  }

  for (int k0 = 0; k0 < EMB; k0 += TK) {
    // ---- commit prefetched tile to LDS (f32 -> bf16)
    __syncthreads();   // previous iteration's fragment reads done
    #pragma unroll
    for (int pass = 0; pass < 4; ++pass) {
      const int r = a_rbase + pass * 32;
      __bf16* ad = &Als[r * LDA + a_kg];
      ad[0] = (__bf16)aprf[pass].x; ad[1] = (__bf16)aprf[pass].y;
      ad[2] = (__bf16)aprf[pass].z; ad[3] = (__bf16)aprf[pass].w;
      const int kk = b_kk0 + pass * 8;
      Bls[(b_cg + 0) * LDB + kk] = (__bf16)bprf[pass].x;
      Bls[(b_cg + 1) * LDB + kk] = (__bf16)bprf[pass].y;
      Bls[(b_cg + 2) * LDB + kk] = (__bf16)bprf[pass].z;
      Bls[(b_cg + 3) * LDB + kk] = (__bf16)bprf[pass].w;
    }
    __syncthreads();

    // ---- prefetch next K-tile from global (overlaps with WMMAs below)
    const int kn = k0 + TK;
    if (kn < EMB) {
      #pragma unroll
      for (int pass = 0; pass < 4; ++pass) {
        aprf[pass] = *(const float4*)(emb + (size_t)tok[a_rbase + pass * 32] * EMB + kn + a_kg);
        bprf[pass] = *(const float4*)(Wx + (size_t)(kn + b_kk0 + pass * 8) * G4 + N0 + b_cg);
      }
    }

    // ---- A fragment: M = lane%16 (both halves), K = e%8 + (e/8)*16 + half*8
    v16bf afrag;
    {
      const __bf16* ap = &Als[(w * 16 + l16) * LDA + half * 8];
      const bf16x8 lo = *(const bf16x8*)(ap);        // K = half*8 + 0..7
      const bf16x8 hi = *(const bf16x8*)(ap + 16);   // K = half*8 + 16..23
      #pragma unroll
      for (int e = 0; e < 8; ++e) { afrag[e] = lo[e]; afrag[8 + e] = hi[e]; }
    }

    // ---- 2-deep software pipeline over the 8 N-tiles:
    // issue ds_loads for fragment j+1 BEFORE the WMMA consuming fragment j,
    // so each WMMA waits only on a load pair issued one stage earlier.
    v16bf bcur = load_bfrag(Bls, 0, l16, half);
    #pragma unroll
    for (int j = 0; j < 8; ++j) {
      v16bf bnext;
      if (j < 7) bnext = load_bfrag(Bls, j + 1, l16, half);
      acc[j] = __builtin_amdgcn_wmma_f32_16x16x32_bf16(
          false, afrag, false, bcur, (short)0, acc[j], false, false);
      bcur = bnext;
    }
  }

  // ---- epilogue: C row = w*16 + half*8 + r, col = j*16 + lane%16; add bias
  #pragma unroll
  for (int j = 0; j < 8; ++j) {
    const int col = N0 + j * 16 + l16;
    const float b = bx[col];
    #pragma unroll
    for (int r = 0; r < 8; ++r) {
      const int row = M0 + w * 16 + half * 8 + r;
      gx[(size_t)row * G4 + col] = acc[j][r] + b;
    }
  }
}

// ---------------------------------------------------------------------------
// Phase 2: persistent sequential LSTM scan. 64 WGs x 256 threads; WG wg owns
// hidden units [wg*16, wg*16+16) => 64 gate columns. Wh stays hot in L2.
// fp32 recurrence; grid barrier (atomic sense counter) between steps.
// ---------------------------------------------------------------------------
#define NWG   64
#define CHUNK 16

__global__ __launch_bounds__(256)
void lstm_scan_kernel(const float* __restrict__ Wh,
                      const float* __restrict__ bh,
                      const float* __restrict__ gx,
                      float*       __restrict__ h_glob,
                      unsigned*    __restrict__ bar,
                      float*       __restrict__ out) {
  __shared__ float hls[HID];
  __shared__ float red[256];
  __shared__ float gl[64];

  const int t    = threadIdx.x;
  const int base = blockIdx.x * CHUNK;
  const int lc   = t & 63;      // local gate column 0..63
  const int p    = t >> 6;      // K partition 0..3
  const int gate = lc >> 4;
  const int idx  = lc & 15;
  const int gcol = gate * HID + base + idx;   // column in [0,4096)
  const float bias = bh[gcol];

  volatile unsigned* cnt = bar;
  volatile unsigned* gen = bar + 1;

  float c = 0.f;   // cell state for hidden unit (base + t), valid t < CHUNK

  for (int step = 0; step < SEQ; ++step) {
    // broadcast h into LDS (4 KB, served from L2)
    *(float4*)(hls + t * 4) = *(const float4*)(h_glob + t * 4);
    __syncthreads();

    // partial GEMV: this thread covers K in [p*256, p*256+256) for one column
    float sum = 0.f;
    const float* wp = Wh + (size_t)(p * 256) * G4 + gcol;
    #pragma unroll 8
    for (int kk = 0; kk < 256; ++kk)
      sum += hls[p * 256 + kk] * wp[(size_t)kk * G4];
    red[t] = sum;
    __syncthreads();

    if (t < 64) {
      gl[lc] = red[lc] + red[64 + lc] + red[128 + lc] + red[192 + lc]
             + bias + gx[(size_t)step * G4 + gcol];
    }
    __syncthreads();

    if (t < CHUNK) {
      const float gi = gl[t], gf = gl[16 + t], go = gl[32 + t], gu = gl[48 + t];
      const float i = 1.f / (1.f + __expf(-gi));
      const float f = 1.f / (1.f + __expf(-gf));
      const float o = 1.f / (1.f + __expf(-go));
      const float u = tanhf(gu);
      c = f * c + i * u;
      const float h = o * tanhf(c);
      h_glob[base + t] = h;
      if (step == SEQ - 1) {
        out[base + t]       = h;   // final h
        out[HID + base + t] = c;   // final c
      }
    }

    // ---- grid barrier (sense via generation counter) ----
    __syncthreads();
    if (t == 0) {
      __threadfence();
      const unsigned g = *gen;
      if (atomicAdd((unsigned*)cnt, 1u) == NWG - 1) {
        atomicExch((unsigned*)cnt, 0u);
        __threadfence();
        atomicAdd((unsigned*)gen, 1u);
      } else {
        while (*gen == g) __builtin_amdgcn_s_sleep(1);
      }
      __threadfence();
    }
    __syncthreads();
  }
}

// ---------------------------------------------------------------------------
extern "C" void kernel_launch(void* const* d_in, const int* in_sizes, int n_in,
                              void* d_out, int out_size, void* d_ws, size_t ws_size,
                              hipStream_t stream) {
  const int*   sent = (const int*)  d_in[0];
  const float* emb  = (const float*)d_in[1];
  const float* Wx   = (const float*)d_in[2];
  const float* bx   = (const float*)d_in[3];
  const float* Wh   = (const float*)d_in[4];
  const float* bh   = (const float*)d_in[5];
  float* out = (float*)d_out;

  char* ws = (char*)d_ws;
  float*    gx     = (float*)ws;                                     // 33.55 MB
  float*    h_glob = (float*)(ws + (size_t)SEQ * G4 * 4);            // 4 KB
  unsigned* bar    = (unsigned*)(ws + (size_t)SEQ * G4 * 4 + HID * 4);

  lstm_init_kernel<<<1, 256, 0, stream>>>(h_glob, bar);
  gx_gemm_kernel<<<dim3(G4 / TN, SEQ / TM), 256, 0, stream>>>(sent, emb, Wx, bx, gx);
  lstm_scan_kernel<<<NWG, 256, 0, stream>>>(Wh, bh, gx, h_glob, bar, out);
}
